// Loss_54769422958908
// MI455X (gfx1250) — matrix operand order repaired and checked by reading, not compiled, not run on hardware
//
#include <hip/hip_runtime.h>

typedef __attribute__((ext_vector_type(16))) _Float16 v16h;
typedef __attribute__((ext_vector_type(8)))  _Float16 v8h;
typedef __attribute__((ext_vector_type(8)))  float    v8f;

#define Bn 8
#define Sn 16
#define Cn 4
#define Hn 160
#define Wn 160
#define HW (Hn*Wn)            /* 25600  */
#define CHW (Cn*HW)           /* 102400 */
#define PR 240                /* padded rows: interior at +48 */
#define PC 224                /* padded cols: interior at +32 */
#define PPLANE (PR*PC)        /* 53760  */
#define PSLICE (Cn*PPLANE)    /* 215040 */
#define NSLICE (Bn*Sn)        /* 128    */
#define NPAIR  (Bn*(Sn-1))    /* 120    */
#define NTILE  9
#define NEG_INF (-3.4e38f)

// ---------------- Kernel 1: supervised term, one block per (b,s) slice -----
__global__ __launch_bounds__(256)
void k_supervised(const float* __restrict__ frames,
                  const float* __restrict__ renders,
                  float* __restrict__ supv) {
  const int sl = blockIdx.x;                       // 0..127
  const float* fb = frames  + (size_t)sl * CHW;
  const float* rb = renders + (size_t)sl * CHW;
  float sAin = 0.f, sAout = 0.f, sM = 0.f, sRGB = 0.f;
  for (int p = threadIdx.x; p < HW; p += blockDim.x) {
    float ga = fb[3*HW + p];
    float ra = rb[3*HW + p];
    float al1 = fabsf(ga - ra);
    float mask = (ga > 0.0f) ? 1.0f : 0.0f;
    float rgb = fabsf(fb[0*HW+p]*ga - rb[0*HW+p]*ra)
              + fabsf(fb[1*HW+p]*ga - rb[1*HW+p]*ra)
              + fabsf(fb[2*HW+p]*ga - rb[2*HW+p]*ra);
    sAin  += al1 * mask;
    sAout += al1 * (1.0f - mask);
    sM    += mask;
    sRGB  += rgb * mask;
  }
  __shared__ float r0[256], r1[256], r2[256], r3[256];
  const int t = threadIdx.x;
  r0[t] = sAin; r1[t] = sAout; r2[t] = sM; r3[t] = sRGB;
  __syncthreads();
  for (int off = 128; off > 0; off >>= 1) {
    if (t < off) { r0[t]+=r0[t+off]; r1[t]+=r1[t+off]; r2[t]+=r2[t+off]; r3[t]+=r3[t+off]; }
    __syncthreads();
  }
  if (t == 0) {
    float m    = r2[0];
    float dIn  = (m == 0.f) ? 1.f : m;               // den + (den==0)
    float mOut = (float)HW - m;
    float dOut = (mOut == 0.f) ? 1.f : mOut;
    supv[sl] = (r0[0]/dIn + r1[0]/dOut + r3[0]/dIn) * (1.0f/3.0f);
  }
}

// ---------------- Kernel 2: per-slice mean / 1/std of renders --------------
__global__ __launch_bounds__(256)
void k_stats(const float* __restrict__ renders,
             float* __restrict__ meanv, float* __restrict__ rstdv) {
  const int sl = blockIdx.x;
  const float* base = renders + (size_t)sl * CHW;
  float s1 = 0.f, s2 = 0.f;
  for (int i = threadIdx.x; i < CHW; i += blockDim.x) {
    float v = base[i]; s1 += v; s2 += v*v;
  }
  __shared__ float a[256], b[256];
  const int t = threadIdx.x;
  a[t] = s1; b[t] = s2; __syncthreads();
  for (int off = 128; off > 0; off >>= 1) {
    if (t < off) { a[t]+=a[t+off]; b[t]+=b[t+off]; }
    __syncthreads();
  }
  if (t == 0) {
    float m   = a[0] * (1.0f/(float)CHW);
    float var = b[0] * (1.0f/(float)CHW) - m*m;
    var = var > 0.f ? var : 0.f;
    float sd = sqrtf(var);
    if (sd == 0.f) sd = 1.f;                         // s + (s==0)
    meanv[sl] = m; rstdv[sl] = 1.f / sd;
  }
}

// ---------------- Kernel 3: normalized f16 frames, zero-padded -------------
__global__ __launch_bounds__(256)
void k_buildP(const float* __restrict__ renders,
              const float* __restrict__ meanv,
              const float* __restrict__ rstdv,
              _Float16* __restrict__ P) {
  long long idx = (long long)blockIdx.x * blockDim.x + threadIdx.x;
  const long long total = (long long)NSLICE * PSLICE;
  if (idx >= total) return;
  int sl  = (int)(idx / PSLICE);
  int rem = (int)(idx % PSLICE);
  int c   = rem / PPLANE;
  int r2  = rem % PPLANE;
  int row = r2 / PC, col = r2 % PC;
  int y = row - 48, x = col - 32;
  _Float16 out = (_Float16)0.0f;
  if ((unsigned)y < (unsigned)Hn && (unsigned)x < (unsigned)Wn) {
    float v = renders[(size_t)sl*CHW + c*HW + y*Wn + x];
    out = (_Float16)((v - meanv[sl]) * rstdv[sl]);
  }
  P[idx] = out;
}

// ---------------- Kernel 4: WMMA ZNCC, one wave per (pair, 16x16 tile) -----
// cc(sy,sx) tile: D[m=sx_local, n=sy_local] = sum_k A[m,k]*B[k,n]
//   A[m,K] = P1[c][yp+32][x0 + K + sx_base + m + 16]   (x-shift in M)
//   B[K,n] = P2[c][yp + 48 - sy_base - n][x0 + K + 32] (y-shift in N)
__global__ __launch_bounds__(32)
void k_zncc(const _Float16* __restrict__ P, float* __restrict__ tcmax) {
  const int tile = blockIdx.x % NTILE;
  const int g    = blockIdx.x / NTILE;               // 0..119
  const int b    = g / (Sn-1), s = g % (Sn-1);
  const _Float16* P1 = P + (size_t)(b*Sn + s) * PSLICE;   // earlier frame (shifted)
  const _Float16* P2 = P1 + PSLICE;                       // later frame (kernel)
  const int tileM = tile % 3, tileN = tile / 3;
  const int lid = threadIdx.x;
  const int lane16 = lid & 15, h = lid >> 4;

  // A-layout (16-bit 16x32): lane holds row M=lane16; elems e -> K=16h+e
  // (matches B)  => two contiguous 16B runs at +0 and +8.
  const int aColBase = tileM*16 + lane16 + 16*h + 16;
  // B-layout used here (32x16): lane holds col N=lane16; elems e -> K=16h+e
  const int bColBase = 16*h + 32;
  const int bRowOff  = 48 - tileN*16 - lane16;       // row = yp + bRowOff in [1,239]

  v8f acc = {0.f,0.f,0.f,0.f,0.f,0.f,0.f,0.f};
  for (int c = 0; c < Cn; ++c) {
    const _Float16* aPlane = P1 + c*PPLANE;
    const _Float16* bPlane = P2 + c*PPLANE;
    for (int yp = 0; yp < 192; ++yp) {
      const _Float16* aRow = aPlane + (yp + 32)   * PC + aColBase;
      const _Float16* bRow = bPlane + (yp + bRowOff)*PC + bColBase;
#pragma unroll
      for (int j = 0; j < 5; ++j) {                  // K chunks of 32 along x
        v8h alo, ahi; v16h bf;
        __builtin_memcpy(&alo, aRow + j*32,      sizeof(alo));
        __builtin_memcpy(&ahi, aRow + j*32 + 8,  sizeof(ahi));
        __builtin_memcpy(&bf,  bRow + j*32,      sizeof(bf));
        v16h af = __builtin_shufflevector(alo, ahi,
                    0,1,2,3,4,5,6,7,8,9,10,11,12,13,14,15);
        acc = __builtin_amdgcn_wmma_f32_16x16x32_f16(
                  false, af, false, bf, (short)0, acc, false, false);
      }
    }
  }

  // D layout: lane(lane16,h), elem j -> (M = j + 8h, N = lane16)
  float mx = NEG_INF;
  const int syi = tileN*16 + lane16;
  if (syi < 33) {
#pragma unroll
    for (int j = 0; j < 8; ++j) {
      int sxi = tileM*16 + j + 8*h;
      if (sxi < 33) mx = fmaxf(mx, acc[j] * (1.0f/(float)CHW));
    }
  }
  for (int off = 16; off > 0; off >>= 1)
    mx = fmaxf(mx, __shfl_xor(mx, off, 32));
  if (lid == 0) tcmax[g*NTILE + tile] = mx;
}

// ---------------- Kernel 5: finalize (8,2) output --------------------------
__global__ __launch_bounds__(32)
void k_final(const float* __restrict__ supv,
             const float* __restrict__ tcmax,
             float* __restrict__ out) {
  const int b = threadIdx.x;
  if (b >= Bn) return;
  float sup = 0.f;
  for (int s = 0; s < Sn; ++s) sup += supv[b*Sn + s];
  sup *= (1.0f/(float)Sn);
  float tacc = 0.f;
  for (int s = 0; s < Sn-1; ++s) {
    float mx = NEG_INF;
    for (int t = 0; t < NTILE; ++t)
      mx = fmaxf(mx, tcmax[(b*(Sn-1)+s)*NTILE + t]);
    tacc += mx;
  }
  tacc *= (1.0f/(float)(Sn-1));
  out[b*2+0] = sup;
  out[b*2+1] = (1.0f - tacc) * 0.5f;
}

// ---------------------------------------------------------------------------
extern "C" void kernel_launch(void* const* d_in, const int* in_sizes, int n_in,
                              void* d_out, int out_size, void* d_ws, size_t ws_size,
                              hipStream_t stream) {
  const float* frames  = (const float*)d_in[0];
  const float* renders = (const float*)d_in[1];
  float* out = (float*)d_out;

  char* ws = (char*)d_ws;
  size_t off = 0;
  _Float16* P = (_Float16*)(ws + off);
  off += (size_t)NSLICE * PSLICE * sizeof(_Float16);
  off = (off + 255) & ~(size_t)255;
  float* supv  = (float*)(ws + off); off += NSLICE * sizeof(float);
  off = (off + 255) & ~(size_t)255;
  float* meanv = (float*)(ws + off); off += NSLICE * sizeof(float);
  off = (off + 255) & ~(size_t)255;
  float* rstdv = (float*)(ws + off); off += NSLICE * sizeof(float);
  off = (off + 255) & ~(size_t)255;
  float* tcmax = (float*)(ws + off); off += (size_t)NPAIR * NTILE * sizeof(float);
  (void)ws_size; (void)in_sizes; (void)n_in; (void)out_size;

  k_supervised<<<NSLICE, 256, 0, stream>>>(frames, renders, supv);
  k_stats     <<<NSLICE, 256, 0, stream>>>(renders, meanv, rstdv);
  const long long totalP = (long long)NSLICE * PSLICE;
  k_buildP<<<(int)((totalP + 255) / 256), 256, 0, stream>>>(renders, meanv, rstdv, P);
  k_zncc  <<<NPAIR * NTILE, 32, 0, stream>>>(P, tcmax);
  k_final <<<1, 32, 0, stream>>>(supv, tcmax, out);
}
